// Attention_64854006169830
// MI455X (gfx1250) — compile-verified, matrix operand order checked by reading, not run on hardware
//
#include <hip/hip_runtime.h>

// ---------------------------------------------------------------------------
// CDNA5 (gfx1250) transformer block: bf16 WMMA GEMMs (async-LDS double
// buffered) + flash attention. Wave32; v_wmma_f32_16x16x32_bf16 everywhere.
// ---------------------------------------------------------------------------

#define DEV __device__ __forceinline__

typedef __attribute__((ext_vector_type(16))) __bf16   bf16x16;
typedef __attribute__((ext_vector_type(8)))  float    f32x8;
typedef __attribute__((ext_vector_type(4)))  unsigned u32x4;

union FragU { u32x4 u[2]; bf16x16 v; };

static constexpr int Bc  = 4;
static constexpr int Lc  = 2048;
static constexpr int Dc  = 1024;
static constexpr int Hc  = 8;
static constexpr int DHc = 128;
static constexpr int OUTc = 256;

// float -> bf16, round-to-nearest-even (bit trick; no exceptions needed)
DEV __bf16 to_bf(float f) {
  union { float f; unsigned u; } a; a.f = f;
  unsigned r = a.u + 0x7FFFu + ((a.u >> 16) & 1u);
  union { unsigned short s; __bf16 b; } o; o.s = (unsigned short)(r >> 16);
  return o.b;
}

// Async copy of one 16B chunk global -> LDS (GLOBAL_LOAD_ASYNC_TO_LDS_B128,
// GV mode: per-lane 32-bit LDS offset in VDST, 64-bit global addr in VADDR).
// Tracked by ASYNCcnt.
DEV void async_g2l_b128(unsigned lds_off, const void* gptr) {
  unsigned long long ga = (unsigned long long)(uintptr_t)gptr;
  asm volatile("global_load_async_to_lds_b128 %0, %1, off"
               :: "v"(lds_off), "v"(ga)
               : "memory");
}
DEV void wait_asynccnt0() {
  asm volatile("s_wait_asynccnt 0x0" ::: "memory");
}

// Load one 16x32 bf16 WMMA A/B fragment from row-major memory.
// tile points at the fragment's first row. Per ISA 7.12.2:
//   lane L (0..15):  row = L,      K = kk+0..7  and kk+16..23
//   lane L (16..31): row = L-16,   K = kk+8..15 and kk+24..31
DEV bf16x16 load_frag(const __bf16* tile, int ld, int kk, int lane) {
  const __bf16* p = tile + (size_t)(lane & 15) * (size_t)ld + kk + ((lane >> 4) << 3);
  FragU f;
  f.u[0] = *(const u32x4*)(p);
  f.u[1] = *(const u32x4*)(p + 16);
  return f.v;
}

// Same fragment pattern out of an LDS tile with leading dim 32 (one K-slab).
DEV bf16x16 lds_frag(const __bf16* tile, int lane) {
  const __bf16* p = tile + (lane & 15) * 32 + ((lane >> 4) << 3);
  FragU f;
  f.u[0] = *(const u32x4*)(p);
  f.u[1] = *(const u32x4*)(p + 16);
  return f.v;
}

DEV f32x8 wmma_bf16(bf16x16 a, bf16x16 b, f32x8 c) {
  return __builtin_amdgcn_wmma_f32_16x16x32_bf16(false, a, false, b, (short)0, c,
                                                 false, false);
}

DEV float hmax16(float v) {
  #pragma unroll
  for (int o = 8; o > 0; o >>= 1) v = fmaxf(v, __shfl_xor(v, o, 16));
  return v;
}
DEV float hsum16(float v) {
  #pragma unroll
  for (int o = 8; o > 0; o >>= 1) v += __shfl_xor(v, o, 16);
  return v;
}

// ---------------------------------------------------------------------------
// Elementwise converts / transposes
// ---------------------------------------------------------------------------
__global__ void cvt_bf16_k(const float* __restrict__ s, __bf16* __restrict__ d, int n) {
  int i = (blockIdx.x * 256 + threadIdx.x) * 4;
  if (i + 3 < n) {
    float4 v = *(const float4*)(s + i);
    d[i + 0] = to_bf(v.x);
    d[i + 1] = to_bf(v.y);
    d[i + 2] = to_bf(v.z);
    d[i + 3] = to_bf(v.w);
  }
}

// W[K,N] f32 -> Wt[N,K] bf16
__global__ void cvt_transpose_k(const float* __restrict__ w, __bf16* __restrict__ wt,
                                int K, int N) {
  int n = blockIdx.x * 16 + (threadIdx.x & 15);
  int k = blockIdx.y * 16 + (threadIdx.x >> 4);
  if (n < N && k < K) wt[(size_t)n * K + k] = to_bf(w[(size_t)k * N + n]);
}

// V bf16 [B,L,D] -> Vt bf16 [B,H,DH,L]
__global__ void transpose_v_k(const __bf16* __restrict__ v, __bf16* __restrict__ vt) {
  int i = blockIdx.x * 256 + threadIdx.x;           // over B*H*DH*L
  int l = i & (Lc - 1);
  int r = i >> 11;
  int d = r & (DHc - 1);
  r >>= 7;
  int h = r & (Hc - 1);
  int b = r >> 3;
  vt[i] = v[((size_t)(b * Lc + l)) * Dc + h * DHc + d];
}

// ---------------------------------------------------------------------------
// bf16 WMMA GEMM with async-LDS double buffering:
//   out = act(A[M,K] @ W[K,N] + bias),  W passed pre-transposed as Wt[N,K].
// Block = 8 waves; block tile 128(M) x 128(N); wave tile 32x64
// (2x4 f32x8 accumulators). Per 32-wide K-slab:
//   global_load_async_to_lds_b128 stages A(128x32) and B(128x32) tiles,
//   s_wait_asynccnt + barrier, ds_load_b128 fragments, 8 WMMAs.
// ---------------------------------------------------------------------------
__global__ __launch_bounds__(256) void gemm_bf16_k(
    const __bf16* __restrict__ A, const __bf16* __restrict__ Wt,
    const float* __restrict__ bias, float* __restrict__ outF,
    __bf16* __restrict__ outBF, int M, int N, int K, int relu) {
  __shared__ alignas(16) __bf16 sA[2][128 * 32];
  __shared__ alignas(16) __bf16 sB[2][128 * 32];

  const int tid  = threadIdx.x;
  const int lane = tid & 31;
  const int w    = tid >> 5;
  const int wr   = w & 3;     // 4 row-groups of 32
  const int wc   = w >> 2;    // 2 col-groups of 64
  const int rowBase = blockIdx.y * 128;
  const int colBase = blockIdx.x * 128;

  // Stage one 32-wide K-slab of A and B tiles into LDS buffer `buf`.
  // 128 rows x 32 cols x 2B = 8KB = 512 16B-chunks per tile; chunk c:
  // row = c>>2, quarter q = c&3 (8 bf16 each).
  auto stage = [&](int buf, int kk) {
    #pragma unroll
    for (int c = tid; c < 512; c += 256) {
      int row = c >> 2, q = c & 3;
      async_g2l_b128((unsigned)(uintptr_t)&sA[buf][row * 32 + q * 8],
                     A + (size_t)(rowBase + row) * K + kk + q * 8);
      async_g2l_b128((unsigned)(uintptr_t)&sB[buf][row * 32 + q * 8],
                     Wt + (size_t)(colBase + row) * K + kk + q * 8);
    }
  };

  f32x8 acc[2][4];
  #pragma unroll
  for (int i = 0; i < 2; ++i)
    #pragma unroll
    for (int j = 0; j < 4; ++j) acc[i][j] = (f32x8){};

  stage(0, 0);
  for (int kk = 0; kk < K; kk += 32) {
    const int buf = (kk >> 5) & 1;
    wait_asynccnt0();          // our async issues have landed in LDS
    __syncthreads();           // everyone's have, and prev compute is done
    if (kk + 32 < K) stage(buf ^ 1, kk + 32);

    bf16x16 a0 = lds_frag(&sA[buf][(wr * 32 + 0) * 32], lane);
    bf16x16 a1 = lds_frag(&sA[buf][(wr * 32 + 16) * 32], lane);
    #pragma unroll
    for (int j = 0; j < 4; ++j) {
      bf16x16 bj = lds_frag(&sB[buf][(wc * 64 + j * 16) * 32], lane);
      acc[0][j] = wmma_bf16(a0, bj, acc[0][j]);
      acc[1][j] = wmma_bf16(a1, bj, acc[1][j]);
    }
  }

  const int hi = lane >> 4, ln = lane & 15;
  #pragma unroll
  for (int ti = 0; ti < 2; ++ti)
    #pragma unroll
    for (int tj = 0; tj < 4; ++tj)
      #pragma unroll
      for (int r = 0; r < 8; ++r) {
        int row = rowBase + wr * 32 + ti * 16 + 8 * hi + r;
        int col = colBase + wc * 64 + tj * 16 + ln;
        float vv = acc[ti][tj][r] + bias[col];
        if (relu) vv = fmaxf(vv, 0.f);
        size_t idx = (size_t)row * N + col;
        if (outF)  outF[idx]  = vv;
        if (outBF) outBF[idx] = to_bf(vv);
      }
}

// ---------------------------------------------------------------------------
// Flash attention: one wave handles 16 query rows of one (b,h); full online
// softmax over L keys in 32-key chunks. Q/K from bf16 [B,L,D] (head slice),
// V pre-transposed bf16 [B,H,DH,L]. ctx written f32 [B,L,D].
// ---------------------------------------------------------------------------
__global__ __launch_bounds__(256) void attn_k(const __bf16* __restrict__ Q,
                                              const __bf16* __restrict__ Km,
                                              const __bf16* __restrict__ Vt,
                                              float* __restrict__ ctx) {
  __shared__ alignas(16) __bf16 sP[8][16 * 32];  // per-wave P staging (C->A relayout)
  const int w    = threadIdx.x >> 5;
  const int lane = threadIdx.x & 31;
  const int bh = blockIdx.y;
  const int b = bh >> 3, h = bh & 7;
  const int q0 = blockIdx.x * 128 + w * 16;

  const __bf16* qb = Q + ((size_t)(b * Lc + q0)) * Dc + h * DHc;
  const __bf16* kb = Km + ((size_t)b * Lc) * Dc + h * DHc;
  const __bf16* vtb = Vt + ((size_t)(b * Hc + h)) * DHc * Lc;

  bf16x16 qa[4];
  #pragma unroll
  for (int i = 0; i < 4; ++i) qa[i] = load_frag(qb, Dc, i * 32, lane);

  f32x8 acc[8];
  #pragma unroll
  for (int t = 0; t < 8; ++t) acc[t] = (f32x8){};
  float mrow[8], lrow[8];
  #pragma unroll
  for (int r = 0; r < 8; ++r) { mrow[r] = -1e30f; lrow[r] = 0.f; }

  const float scale = 0.08838834764831845f;  // 1/sqrt(128)
  const int hi = lane >> 4, ln = lane & 15;

  for (int j0 = 0; j0 < Lc; j0 += 32) {
    // S = Q @ K^T for 32 key columns (two 16x16 f32 tiles)
    f32x8 s0 = {}, s1 = {};
    #pragma unroll
    for (int i = 0; i < 4; ++i) {
      bf16x16 k0 = load_frag(kb + (size_t)j0 * Dc, Dc, i * 32, lane);
      bf16x16 k1 = load_frag(kb + (size_t)(j0 + 16) * Dc, Dc, i * 32, lane);
      s0 = wmma_bf16(qa[i], k0, s0);
      s1 = wmma_bf16(qa[i], k1, s1);
    }
    // online softmax update (row stats via half-wave reductions)
    float p0[8], p1[8], alpha[8];
    #pragma unroll
    for (int r = 0; r < 8; ++r) {
      float v0 = s0[r] * scale, v1 = s1[r] * scale;
      float mx = hmax16(fmaxf(v0, v1));
      float mnew = fmaxf(mrow[r], mx);
      alpha[r] = __expf(mrow[r] - mnew);
      p0[r] = __expf(v0 - mnew);
      p1[r] = __expf(v1 - mnew);
      float rs = hsum16(p0[r] + p1[r]);
      lrow[r] = lrow[r] * alpha[r] + rs;
      mrow[r] = mnew;
    }
    #pragma unroll
    for (int t = 0; t < 8; ++t)
      #pragma unroll
      for (int r = 0; r < 8; ++r) acc[t][r] *= alpha[r];

    // C-layout P tile -> LDS -> A-layout fragment
    #pragma unroll
    for (int r = 0; r < 8; ++r) {
      sP[w][(8 * hi + r) * 32 + ln]      = to_bf(p0[r]);
      sP[w][(8 * hi + r) * 32 + 16 + ln] = to_bf(p1[r]);
    }
    bf16x16 pf = lds_frag(&sP[w][0], lane);
    // acc += P(16x32) @ V(32x128): one WMMA per 16-wide DH chunk
    #pragma unroll
    for (int t = 0; t < 8; ++t) {
      bf16x16 bv = load_frag(vtb + (size_t)(t * 16) * Lc, Lc, j0, lane);
      acc[t] = wmma_bf16(pf, bv, acc[t]);
    }
  }

  #pragma unroll
  for (int t = 0; t < 8; ++t)
    #pragma unroll
    for (int r = 0; r < 8; ++r) {
      int row = q0 + 8 * hi + r;
      int col = h * DHc + t * 16 + ln;
      ctx[((size_t)(b * Lc + row)) * Dc + col] = acc[t][r] / lrow[r];
    }
}

// ---------------------------------------------------------------------------
// Fused residual + LayerNorm over D=1024; writes f32 (optional) + bf16.
// One 256-thread block per row.
// ---------------------------------------------------------------------------
__global__ __launch_bounds__(256) void ln_k(const float* __restrict__ A,
                                            const float* __restrict__ Bs,
                                            const float* __restrict__ gamma,
                                            const float* __restrict__ beta,
                                            float* __restrict__ outF,
                                            __bf16* __restrict__ outBF) {
  __shared__ float red[16];
  const int row = blockIdx.x, t = threadIdx.x;
  float v[4], s = 0.f, ss = 0.f;
  #pragma unroll
  for (int i = 0; i < 4; ++i) {
    int idx = t + i * 256;
    float xv = A[(size_t)row * Dc + idx] + Bs[(size_t)row * Dc + idx];
    v[i] = xv; s += xv; ss += xv * xv;
  }
  #pragma unroll
  for (int o = 16; o > 0; o >>= 1) { s += __shfl_xor(s, o, 32); ss += __shfl_xor(ss, o, 32); }
  if ((t & 31) == 0) { red[t >> 5] = s; red[8 + (t >> 5)] = ss; }
  __syncthreads();
  float S = 0.f, SS = 0.f;
  #pragma unroll
  for (int i = 0; i < 8; ++i) { S += red[i]; SS += red[8 + i]; }
  float mean = S * (1.0f / Dc);
  float var = SS * (1.0f / Dc) - mean * mean;
  float rstd = rsqrtf(var + 1e-12f);
  #pragma unroll
  for (int i = 0; i < 4; ++i) {
    int idx = t + i * 256;
    float y = gamma[idx] * (v[i] - mean) * rstd + beta[idx];
    size_t o = (size_t)row * Dc + idx;
    if (outF) outF[o] = y;
    outBF[o] = to_bf(y);
  }
}

// relu(C[row,0:1024] . wc2 + bc2) -> cvec[row]; one block per row
__global__ __launch_bounds__(256) void c2_k(const float* __restrict__ C,
                                            const float* __restrict__ wc2,
                                            const float* __restrict__ bc2,
                                            float* __restrict__ cvec) {
  __shared__ float red[8];
  const int row = blockIdx.x, t = threadIdx.x;
  float s = 0.f;
  #pragma unroll
  for (int i = 0; i < 4; ++i) {
    int idx = t + i * 256;
    s += C[(size_t)row * Dc + idx] * wc2[idx];
  }
  #pragma unroll
  for (int o = 16; o > 0; o >>= 1) s += __shfl_xor(s, o, 32);
  if ((t & 31) == 0) red[t >> 5] = s;
  __syncthreads();
  if (t == 0) {
    float S = 0.f;
    #pragma unroll
    for (int i = 0; i < 8; ++i) S += red[i];
    cvec[row] = fmaxf(S + bc2[0], 0.f);
  }
}

// relu(cvec[B,L] @ wl1[L,L] + bl1) -> cl1[B,L]
__global__ void l1_k(const float* __restrict__ cvec, const float* __restrict__ wl1,
                     const float* __restrict__ bl1, float* __restrict__ cl1) {
  const int j = blockIdx.x * 256 + threadIdx.x;
  const int b = blockIdx.y;
  float s = bl1[j];
  for (int l = 0; l < Lc; ++l) s += cvec[b * Lc + l] * wl1[(size_t)l * Lc + j];
  cl1[b * Lc + j] = fmaxf(s, 0.f);
}

// cl1[B,L] @ wl2[L,OUT] + bl2 -> out[B,OUT]
__global__ void l2_k(const float* __restrict__ cl1, const float* __restrict__ wl2,
                     const float* __restrict__ bl2, float* __restrict__ out) {
  const int o = threadIdx.x;
  const int b = blockIdx.x;
  float s = bl2[o];
  for (int l = 0; l < Lc; ++l) s += cl1[b * Lc + l] * wl2[(size_t)l * OUTc + o];
  out[b * OUTc + o] = s;
}

// ---------------------------------------------------------------------------
extern "C" void kernel_launch(void* const* d_in, const int* in_sizes, int n_in,
                              void* d_out, int out_size, void* d_ws, size_t ws_size,
                              hipStream_t stream) {
  const float* x    = (const float*)d_in[0];
  const float* wq   = (const float*)d_in[1];
  const float* bq   = (const float*)d_in[2];
  const float* wk   = (const float*)d_in[3];
  const float* bk   = (const float*)d_in[4];
  const float* wv   = (const float*)d_in[5];
  const float* bv   = (const float*)d_in[6];
  const float* gamma= (const float*)d_in[7];
  const float* beta = (const float*)d_in[8];
  const float* wff  = (const float*)d_in[9];
  const float* bff  = (const float*)d_in[10];
  const float* wc1  = (const float*)d_in[11];
  const float* bc1  = (const float*)d_in[12];
  const float* wc2  = (const float*)d_in[13];
  const float* bc2  = (const float*)d_in[14];
  const float* wl1  = (const float*)d_in[15];
  const float* bl1  = (const float*)d_in[16];
  const float* wl2  = (const float*)d_in[17];
  const float* bl2  = (const float*)d_in[18];
  float* out = (float*)d_out;

  const int ROWS = Bc * Lc;           // 8192
  const size_t nXD = (size_t)ROWS * Dc;

  char* ws = (char*)d_ws;
  size_t off = 0;
  auto alloc = [&](size_t bytes) {
    size_t o = off;
    off += (bytes + 255) & ~(size_t)255;
    return o;
  };
  __bf16* xbf  = (__bf16*)(ws + alloc(nXD * 2));
  __bf16* wqT  = (__bf16*)(ws + alloc((size_t)Dc * Dc * 2));
  __bf16* wkT  = (__bf16*)(ws + alloc((size_t)Dc * Dc * 2));
  __bf16* wvT  = (__bf16*)(ws + alloc((size_t)Dc * Dc * 2));
  __bf16* wffT = (__bf16*)(ws + alloc((size_t)Dc * Dc * 2));
  __bf16* wc1T = (__bf16*)(ws + alloc((size_t)Dc * Dc * 2));
  __bf16* qbf  = (__bf16*)(ws + alloc(nXD * 2));
  __bf16* kbf  = (__bf16*)(ws + alloc(nXD * 2));
  __bf16* vbf  = (__bf16*)(ws + alloc(nXD * 2));
  __bf16* vT   = (__bf16*)(ws + alloc(nXD * 2));
  float*  ctx  = (float*)(ws + alloc(nXD * 4));
  float*  h1f  = (float*)(ws + alloc(nXD * 4));
  __bf16* h1b  = (__bf16*)(ws + alloc(nXD * 2));
  float*  fff  = (float*)(ws + alloc(nXD * 4));
  __bf16* h2b  = (__bf16*)(ws + alloc(nXD * 2));
  float*  cf   = (float*)(ws + alloc(nXD * 4));
  float*  cvec = (float*)(ws + alloc((size_t)ROWS * 4));
  float*  cl1  = (float*)(ws + alloc((size_t)ROWS * 4));
  (void)ws_size; (void)in_sizes; (void)n_in; (void)out_size;

  // 1) fp32 -> bf16 conversions
  cvt_bf16_k<<<(int)(nXD / 1024), 256, 0, stream>>>(x, xbf, (int)nXD);
  dim3 tg(Dc / 16, Dc / 16);
  cvt_transpose_k<<<tg, 256, 0, stream>>>(wq, wqT, Dc, Dc);
  cvt_transpose_k<<<tg, 256, 0, stream>>>(wk, wkT, Dc, Dc);
  cvt_transpose_k<<<tg, 256, 0, stream>>>(wv, wvT, Dc, Dc);
  cvt_transpose_k<<<tg, 256, 0, stream>>>(wff, wffT, Dc, Dc);
  cvt_transpose_k<<<tg, 256, 0, stream>>>(wc1, wc1T, Dc, Dc);

  // 2) QKV projections (bf16 out only)
  dim3 gg(Dc / 128, ROWS / 128);
  gemm_bf16_k<<<gg, 256, 0, stream>>>(xbf, wqT, bq, nullptr, qbf, ROWS, Dc, Dc, 0);
  gemm_bf16_k<<<gg, 256, 0, stream>>>(xbf, wkT, bk, nullptr, kbf, ROWS, Dc, Dc, 0);
  gemm_bf16_k<<<gg, 256, 0, stream>>>(xbf, wvT, bv, nullptr, vbf, ROWS, Dc, Dc, 0);

  // 3) V -> [B,H,DH,L]
  transpose_v_k<<<(int)(nXD / 256), 256, 0, stream>>>(vbf, vT);

  // 4) flash attention -> ctx (f32)
  attn_k<<<dim3(Lc / 128, Bc * Hc), 256, 0, stream>>>(qbf, kbf, vT, ctx);

  // 5) h1 = LN(x + ctx)
  ln_k<<<ROWS, 256, 0, stream>>>(x, ctx, gamma, beta, h1f, h1b);

  // 6) ff = relu(h1 @ wff + bff)
  gemm_bf16_k<<<gg, 256, 0, stream>>>(h1b, wffT, bff, fff, nullptr, ROWS, Dc, Dc, 1);

  // 7) h2 = LN(h1 + ff)  (bf16 only needed downstream)
  ln_k<<<ROWS, 256, 0, stream>>>(h1f, fff, gamma, beta, nullptr, h2b);

  // 8) c = relu(h2 @ wc1 + bc1)
  gemm_bf16_k<<<gg, 256, 0, stream>>>(h2b, wc1T, bc1, cf, nullptr, ROWS, Dc, Dc, 1);

  // 9) collapse D -> 1
  c2_k<<<ROWS, 256, 0, stream>>>(cf, wc2, bc2, cvec);

  // 10) collapse L -> L (relu), then L -> OUT
  l1_k<<<dim3(Lc / 256, Bc), 256, 0, stream>>>(cvec, wl1, bl1, cl1);
  l2_k<<<Bc, OUTc, 0, stream>>>(cl1, wl2, bl2, out);
}
